// SVDQW4A4Linear_24627342475386
// MI455X (gfx1250) — matrix operand order repaired and checked by reading, not compiled, not run on hardware
//
#include <hip/hip_runtime.h>

// ---------------------------------------------------------------------------
// SVDQuant-style W4A4 linear for MI455X (gfx1250, wave32, WMMA)
//   out = x @ dequant(qweight, wscales)^T + (x @ proj_down^T) @ proj_up^T * wtscale + bias
// bf16 WMMA / f32 accum, 128x128 tiles, BK=64 double-buffered LDS staging,
// LoRA folded in as one extra WMMA k-step.
//   * native bf16 conversion ((__bf16) casts -> v_cvt_pk_bf16_f32)
//   * dequant as one i32->f32 cvt + fma(q, s, -8s)
//   * qweight tiles staged with GLOBAL_LOAD_ASYNC_TO_LDS_B128 (ASYNCcnt);
//     builtin signature is (v4i AS1* src, v4i AS3* dst, imm off, imm cpol)
// ---------------------------------------------------------------------------

#define M_DIM 8192
#define K_DIM 4096
#define N_DIM 4096
#define R_DIM 32

typedef __attribute__((ext_vector_type(2)))  __bf16 bf16x2;
typedef __attribute__((ext_vector_type(4)))  __bf16 bf16x4;
typedef __attribute__((ext_vector_type(8)))  __bf16 bf16x8;
typedef __attribute__((ext_vector_type(16))) __bf16 bf16x16;
typedef __attribute__((ext_vector_type(8)))  float  f32x8;
typedef __attribute__((ext_vector_type(4)))  int    i32x4;

#if __has_builtin(__builtin_amdgcn_global_load_async_to_lds_b128) && \
    __has_builtin(__builtin_amdgcn_s_wait_asynccnt)
#define USE_ASYNC_QW 1
#define AS1 __attribute__((address_space(1)))
#define AS3 __attribute__((address_space(3)))
#else
#define USE_ASYNC_QW 0
#endif

static __device__ __forceinline__ bf16x4 cvt4(float a, float b, float c, float d) {
    bf16x4 r;
    r[0] = (__bf16)a; r[1] = (__bf16)b; r[2] = (__bf16)c; r[3] = (__bf16)d;
    return r;
}

// A fragment (16x32 bf16, MxK): lane ln holds row M=ln; lanes 0-15 take K
// kb+[0..7] (elems 0..7) and kb+[16..23] (elems 8..15); lanes 16-31 take
// kb+[8..15] and kb+[24..31].
static __device__ __forceinline__ bf16x16
frag_a(const __bf16* base, int ld, int row, int kb, int laneHi) {
    const __bf16* p = base + row * ld + kb + laneHi * 8;
    bf16x8 lo = *(const bf16x8*)p;          // ds_load_b128
    bf16x8 hi = *(const bf16x8*)(p + 16);   // ds_load_b128
    bf16x16 r;
    #pragma unroll
    for (int i = 0; i < 8; ++i) { r[i] = lo[i]; r[i + 8] = hi[i]; }
    return r;
}

// B fragment (32x16 bf16, KxN): lane ln holds column N=ln; lanes 0-15 hold
// K kb+[0..7] and kb+[16..23], lanes 16-31 hold K kb+[8..15] and kb+[24..31].
static __device__ __forceinline__ bf16x16
frag_b(const __bf16* base, int ld, int row, int kb, int laneHi) {
    const __bf16* p = base + row * ld + kb + laneHi * 8;
    bf16x8 lo = *(const bf16x8*)p;
    bf16x8 hi = *(const bf16x8*)(p + 16);
    bf16x16 r;
    #pragma unroll
    for (int i = 0; i < 8; ++i) { r[i] = lo[i]; r[i + 8] = hi[i]; }
    return r;
}

// ---------------------------------------------------------------------------
// Kernel 1: t[M, R] = x @ proj_down^T   (bf16 WMMA, R = 32 -> 2 N-tiles)
// ---------------------------------------------------------------------------
__global__ __launch_bounds__(256)
void lora_down_kernel(const float* __restrict__ x,
                      const float* __restrict__ pd,
                      float* __restrict__ t) {
    constexpr int BM = 128, BK = 64, NT = 256;
    __shared__ __bf16 Xs[2 * BM * BK];      // 32 KB
    __shared__ __bf16 Ps[2 * R_DIM * BK];   // 8 KB

    const int tid = threadIdx.x;
    const int lane = tid & 31, wv = tid >> 5;
    const int ln = lane & 15, laneHi = lane >> 4;
    const int bm0 = blockIdx.x * BM;

    f32x8 acc[2] = { {}, {} };
    float4 xv[8], pv[2];

    auto issue = [&](int k0) {
        #pragma unroll
        for (int i = 0; i < 8; ++i) {
            int idx = tid + i * NT;                 // 2048 float4 of x tile
            int row = idx >> 4, kq = idx & 15;
            xv[i] = *(const float4*)(x + (size_t)(bm0 + row) * K_DIM + k0 + kq * 4);
        }
        #pragma unroll
        for (int i = 0; i < 2; ++i) {
            int idx = tid + i * NT;                 // 512 float4 of Pd tile
            int row = idx >> 4, kq = idx & 15;
            pv[i] = *(const float4*)(pd + (size_t)row * K_DIM + k0 + kq * 4);
        }
    };
    auto convert = [&](int buf) {
        __bf16* X = Xs + buf * BM * BK;
        __bf16* P = Ps + buf * R_DIM * BK;
        #pragma unroll
        for (int i = 0; i < 8; ++i) {
            int idx = tid + i * NT;
            int row = idx >> 4, kq = idx & 15;
            *(bf16x4*)(X + row * BK + kq * 4) = cvt4(xv[i].x, xv[i].y, xv[i].z, xv[i].w);
        }
        #pragma unroll
        for (int i = 0; i < 2; ++i) {
            int idx = tid + i * NT;
            int row = idx >> 4, kq = idx & 15;
            *(bf16x4*)(P + row * BK + kq * 4) = cvt4(pv[i].x, pv[i].y, pv[i].z, pv[i].w);
        }
    };
    auto compute = [&](int buf) {
        const __bf16* X = Xs + buf * BM * BK;
        const __bf16* P = Ps + buf * R_DIM * BK;
        #pragma unroll
        for (int ks = 0; ks < BK; ks += 32) {
            bf16x16 afr = frag_a(X, BK, wv * 16 + ln, ks, laneHi);
            #pragma unroll
            for (int nt = 0; nt < 2; ++nt) {
                bf16x16 bfr = frag_b(P, BK, nt * 16 + ln, ks, laneHi);
                acc[nt] = __builtin_amdgcn_wmma_f32_16x16x32_bf16(
                    false, afr, false, bfr, (short)0, acc[nt], false, false);
            }
        }
    };

    const int NB = K_DIM / BK;
    issue(0);
    #pragma unroll 1
    for (int blk = 0; blk < NB; ++blk) {
        int buf = blk & 1;
        convert(buf);
        __syncthreads();
        if (blk + 1 < NB) issue((blk + 1) * BK);
        compute(buf);
    }

    #pragma unroll
    for (int nt = 0; nt < 2; ++nt) {
        int col = nt * 16 + ln;
        int row0 = bm0 + wv * 16 + 8 * laneHi;
        #pragma unroll
        for (int j = 0; j < 8; ++j)
            t[(size_t)(row0 + j) * R_DIM + col] = acc[nt][j];
    }
}

// ---------------------------------------------------------------------------
// Kernel 2: out = x @ dequant(W)^T + t @ (wtscale*Pu)^T + bias
// ---------------------------------------------------------------------------
__global__ __launch_bounds__(256)
void w4a4_gemm_kernel(const float* __restrict__ x,
                      const int*   __restrict__ qw,
                      const float* __restrict__ wscales,
                      const float* __restrict__ tlora,
                      const float* __restrict__ proj_up,
                      const float* __restrict__ wtscale,
                      const float* __restrict__ bias,
                      float* __restrict__ out) {
    constexpr int BM = 128, BN = 128, BK = 64, NT = 256;
    __shared__ __bf16 As[2 * BM * BK];      // 32 KB (bf16 x tile)
    __shared__ __bf16 Bs[2 * BN * BK];      // 32 KB (bf16 dequant W tile)
#if USE_ASYNC_QW
    __shared__ int Qraw[2 * BN * BK / 2];   // 32 KB raw packed-int4 staging
#endif

    const int tid = threadIdx.x;
    const int lane = tid & 31, wv = tid >> 5;
    const int ln = lane & 15, laneHi = lane >> 4;
    const int waveM = wv >> 2, waveN = wv & 3;      // 2 x 4 wave grid
    const int bm0 = blockIdx.y * BM;
    const int bn0 = blockIdx.x * BN;

    f32x8 acc[4][2];
    #pragma unroll
    for (int mt = 0; mt < 4; ++mt)
        #pragma unroll
        for (int nt = 0; nt < 2; ++nt) acc[mt][nt] = {};

    float4 xv[8]; float sv[4];
#if !USE_ASYNC_QW
    int4 qv[4];
#endif

    // ---- staging: global -> regs (x, scales) + async/reg (qweight) ---------
    auto issue_x = [&](int k0) {
        #pragma unroll
        for (int i = 0; i < 8; ++i) {               // 2048 float4 of x tile
            int idx = tid + i * NT;
            int row = idx >> 4, kq = idx & 15;
            const float* p = x + (size_t)(bm0 + row) * K_DIM + k0 + kq * 4;
            xv[i] = *(const float4*)p;
            if (k0 + BK < K_DIM) __builtin_prefetch(p + 2 * BK, 0, 1); // distance-2
        }
        const int g = k0 >> 6;                      // GROUP_SIZE = 64 = BK
        #pragma unroll
        for (int i = 0; i < 4; ++i) {
            int idx = tid + i * NT;
            int row = idx >> 3;
            sv[i] = wscales[(size_t)g * N_DIM + bn0 + row];
        }
    };
#if USE_ASYNC_QW
    auto issue_q = [&](int k0, int buf) {
        int* qbase = Qraw + buf * (BN * BK / 2);
        #pragma unroll
        for (int i = 0; i < 4; ++i) {               // b128 async global->LDS copies
            int idx = tid + i * NT;
            int row = idx >> 3, cq = idx & 7;
            const int* src = qw + (size_t)(bn0 + row) * (K_DIM / 2) + (k0 >> 1) + cq * 4;
            __builtin_amdgcn_global_load_async_to_lds_b128(
                (AS1 i32x4*)src, (AS3 i32x4*)(qbase + idx * 4), 0, 0);
        }
    };
#else
    auto issue_q = [&](int k0, int) {
        #pragma unroll
        for (int i = 0; i < 4; ++i) {
            int idx = tid + i * NT;
            int row = idx >> 3, cq = idx & 7;
            qv[i] = *(const int4*)(qw + (size_t)(bn0 + row) * (K_DIM / 2) + (k0 >> 1) + cq * 4);
        }
    };
#endif

    // ---- staging: regs/raw-LDS -> bf16 tiles in LDS ------------------------
    auto convert = [&](int buf) {
        __bf16* A = As + buf * BM * BK;
        __bf16* B = Bs + buf * BN * BK;
        #pragma unroll
        for (int i = 0; i < 8; ++i) {
            int idx = tid + i * NT;
            int row = idx >> 4, kq = idx & 15;
            *(bf16x4*)(A + row * BK + kq * 4) = cvt4(xv[i].x, xv[i].y, xv[i].z, xv[i].w);
        }
        #pragma unroll
        for (int i = 0; i < 4; ++i) {
            int idx = tid + i * NT;
            int row = idx >> 3, cq = idx & 7;
            float s  = sv[i];
            float b8 = s * -8.0f;                   // (q-8)*s == fma(q, s, -8s)
#if USE_ASYNC_QW
            int4 q4 = *(const int4*)(Qraw + buf * (BN * BK / 2) + idx * 4);
#else
            int4 q4 = qv[i];
#endif
            __bf16* d = B + row * BK + cq * 8;
            int q0 = q4.x, q1 = q4.y, q2 = q4.z, q3 = q4.w;
            *(bf16x4*)(d) = cvt4(
                fmaf((float)(q0 & 15), s, b8), fmaf((float)((q0 >> 4) & 15), s, b8),
                fmaf((float)(q1 & 15), s, b8), fmaf((float)((q1 >> 4) & 15), s, b8));
            *(bf16x4*)(d + 4) = cvt4(
                fmaf((float)(q2 & 15), s, b8), fmaf((float)((q2 >> 4) & 15), s, b8),
                fmaf((float)(q3 & 15), s, b8), fmaf((float)((q3 >> 4) & 15), s, b8));
        }
    };

    auto compute = [&](int buf) {
        const __bf16* A = As + buf * BM * BK;
        const __bf16* B = Bs + buf * BN * BK;
        #pragma unroll
        for (int ks = 0; ks < BK; ks += 32) {
            bf16x16 bfr[2];
            #pragma unroll
            for (int nt = 0; nt < 2; ++nt)
                bfr[nt] = frag_b(B, BK, waveN * 32 + nt * 16 + ln, ks, laneHi);
            #pragma unroll
            for (int mt = 0; mt < 4; ++mt) {
                bf16x16 afr = frag_a(A, BK, waveM * 64 + mt * 16 + ln, ks, laneHi);
                #pragma unroll
                for (int nt = 0; nt < 2; ++nt)
                    acc[mt][nt] = __builtin_amdgcn_wmma_f32_16x16x32_bf16(
                        false, afr, false, bfr[nt], (short)0, acc[mt][nt], false, false);
            }
        }
    };

    // ---- main loop ---------------------------------------------------------
    const int NB = K_DIM / BK;
    issue_x(0);
    issue_q(0, 0);
    #pragma unroll 1
    for (int blk = 0; blk < NB; ++blk) {
        int buf = blk & 1;
#if USE_ASYNC_QW
        __builtin_amdgcn_s_wait_asynccnt(0);        // raw q tile resident
#endif
        convert(buf);
        __syncthreads();
        if (blk + 1 < NB) {
            issue_x((blk + 1) * BK);
            issue_q((blk + 1) * BK, buf ^ 1);       // overlaps WMMA burst
        }
        compute(buf);
    }

    // ---- LoRA epilogue: one extra WMMA k-step, K = R = 32 ------------------
    __syncthreads();
    {
        const float ws = wtscale[0];
        #pragma unroll
        for (int i = 0; i < 4; ++i) {               // t tile * wtscale -> As
            int idx = tid + i * NT;
            int row = idx >> 3, rq = idx & 7;
            float4 v = *(const float4*)(tlora + (size_t)(bm0 + row) * R_DIM + rq * 4);
            *(bf16x4*)(As + row * R_DIM + rq * 4) =
                cvt4(v.x * ws, v.y * ws, v.z * ws, v.w * ws);
        }
        #pragma unroll
        for (int i = 0; i < 4; ++i) {               // proj_up tile -> Bs
            int idx = tid + i * NT;
            int row = idx >> 3, rq = idx & 7;
            float4 v = *(const float4*)(proj_up + (size_t)(bn0 + row) * R_DIM + rq * 4);
            *(bf16x4*)(Bs + row * R_DIM + rq * 4) = cvt4(v.x, v.y, v.z, v.w);
        }
    }
    __syncthreads();
    {
        bf16x16 bfr[2];
        #pragma unroll
        for (int nt = 0; nt < 2; ++nt)
            bfr[nt] = frag_b(Bs, R_DIM, waveN * 32 + nt * 16 + ln, 0, laneHi);
        #pragma unroll
        for (int mt = 0; mt < 4; ++mt) {
            bf16x16 afr = frag_a(As, R_DIM, waveM * 64 + mt * 16 + ln, 0, laneHi);
            #pragma unroll
            for (int nt = 0; nt < 2; ++nt)
                acc[mt][nt] = __builtin_amdgcn_wmma_f32_16x16x32_bf16(
                    false, afr, false, bfr[nt], (short)0, acc[mt][nt], false, false);
        }
    }

    // ---- store: C/D layout -> VGPR j of lane l is (M = j + 8*(l>=16), N = l&15)
    #pragma unroll
    for (int nt = 0; nt < 2; ++nt) {
        int col = bn0 + waveN * 32 + nt * 16 + ln;
        float b = bias[col];
        #pragma unroll
        for (int mt = 0; mt < 4; ++mt) {
            int row0 = bm0 + waveM * 64 + mt * 16 + 8 * laneHi;
            #pragma unroll
            for (int j = 0; j < 8; ++j)
                out[(size_t)(row0 + j) * N_DIM + col] = acc[mt][nt][j] + b;
        }
    }
}

// ---------------------------------------------------------------------------
extern "C" void kernel_launch(void* const* d_in, const int* in_sizes, int n_in,
                              void* d_out, int out_size, void* d_ws, size_t ws_size,
                              hipStream_t stream) {
    const float* x   = (const float*)d_in[0];
    const int*   qw  = (const int*)d_in[1];
    const float* wsc = (const float*)d_in[2];
    const float* pd  = (const float*)d_in[3];
    const float* pu  = (const float*)d_in[4];
    const float* wts = (const float*)d_in[5];
    const float* bia = (const float*)d_in[6];
    float* out = (float*)d_out;
    float* t   = (float*)d_ws;   // [M, R] f32 = 1 MB scratch

    lora_down_kernel<<<M_DIM / 128, 256, 0, stream>>>(x, pd, t);

    dim3 grid(N_DIM / 128, M_DIM / 128);
    w4a4_gemm_kernel<<<grid, 256, 0, stream>>>(x, qw, wsc, t, pu, wts, bia, out);
}
